// DiffusionTransformer_2207613190422
// MI455X (gfx1250) — matrix operand order repaired and checked by reading, not compiled, not run on hardware
//
#include <hip/hip_runtime.h>
#include <hip/hip_bf16.h>

// Problem constants (match reference)
#define DMODEL 512
#define NHEAD  8
#define NLAYER 8
#define NRBF   16
#define DFFN   2048
#define BATCH  4
#define SEQL   512
#define HDIM   64   // DMODEL / NHEAD

typedef _Float16 v16h __attribute__((ext_vector_type(16)));
typedef _Float16 v8h  __attribute__((ext_vector_type(8)));
typedef _Float16 v4h  __attribute__((ext_vector_type(4)));
typedef _Float16 v2h  __attribute__((ext_vector_type(2)));
typedef float    v8f  __attribute__((ext_vector_type(8)));

__device__ __forceinline__ float gelu_exact(float x) {
    return 0.5f * x * (1.0f + erff(x * 0.70710678118654752f));
}

// ---------------------------------------------------------------------------
// Small / elementwise kernels (K too small for WMMA to pay off)
// ---------------------------------------------------------------------------

__global__ __launch_bounds__(256) void dit_sinus_t_kernel(const int* __restrict__ t,
                                                          float* __restrict__ out) {
    int idx = blockIdx.x * blockDim.x + threadIdx.x;
    if (idx >= BATCH * DMODEL) return;
    int b = idx / DMODEL, d = idx % DMODEL;
    int j = d >> 1;
    float freq = expf(-logf(10000.0f) * (float)j / 256.0f);
    float ang = (float)t[b] * freq;
    out[idx] = (d & 1) ? cosf(ang) : sinf(ang);
}

// Plain per-thread GEMM for tiny shapes: time-MLP (M=4) and final head (N=3).
__global__ __launch_bounds__(256) void dit_plain_gemm_kernel(
    const float* __restrict__ A, const float* __restrict__ W,
    const float* __restrict__ bias, float* __restrict__ D,
    int M, int N, int K, int do_gelu) {
    int idx = blockIdx.x * blockDim.x + threadIdx.x;
    if (idx >= M * N) return;
    int m = idx / N, n = idx % N;
    float acc = bias ? bias[n] : 0.0f;
    for (int k = 0; k < K; ++k)
        acc = fmaf(A[(size_t)m * K + k], W[(size_t)k * N + n], acc);
    if (do_gelu) acc = gelu_exact(acc);
    D[idx] = acc;
}

// h = x_t@Wi + bi + xsc@Wsc + bsc + pos(l) + temb(b)   (K=3 -> VALU)
__global__ __launch_bounds__(256) void dit_embed_kernel(
    const float* __restrict__ x_t, const float* __restrict__ xsc,
    const float* __restrict__ Wi, const float* __restrict__ bi,
    const float* __restrict__ Wsc, const float* __restrict__ bsc,
    const float* __restrict__ temb, float* __restrict__ h) {
    int idx = blockIdx.x * blockDim.x + threadIdx.x;
    if (idx >= BATCH * SEQL * DMODEL) return;
    int d  = idx % DMODEL;
    int bl = idx / DMODEL;
    int l  = bl % SEQL;
    int b  = bl / SEQL;
    const float* xa = x_t + (size_t)bl * 3;
    const float* xb = xsc + (size_t)bl * 3;
    float acc = bi[d] + bsc[d];
    #pragma unroll
    for (int c = 0; c < 3; ++c)
        acc += xa[c] * Wi[c * DMODEL + d] + xb[c] * Wsc[c * DMODEL + d];
    int j = d >> 1;
    float freq = expf(-logf(10000.0f) * (float)j / 256.0f);
    float ang = (float)l * freq;
    acc += (d & 1) ? cosf(ang) : sinf(ang);
    acc += temb[b * DMODEL + d];
    h[idx] = acc;
}

// rbf[b,l,m,r]: computed once, reused by all 8 layers x 8 heads out of L2.
__global__ __launch_bounds__(256) void dit_rbf_kernel(const float* __restrict__ x_t,
                                                      float* __restrict__ rbf) {
    int idx = blockIdx.x * blockDim.x + threadIdx.x;
    if (idx >= BATCH * SEQL * SEQL) return;
    int m  = idx % SEQL;
    int t2 = idx / SEQL;
    int l  = t2 % SEQL;
    int b  = t2 / SEQL;
    const float* pl = x_t + ((size_t)b * SEQL + l) * 3;
    const float* pm = x_t + ((size_t)b * SEQL + m) * 3;
    float dx = pl[0] - pm[0], dy = pl[1] - pm[1], dz = pl[2] - pm[2];
    float dist = sqrtf(fmaxf(dx * dx + dy * dy + dz * dz, 1e-12f));
    float* o = rbf + (size_t)idx * NRBF;
    #pragma unroll
    for (int r = 0; r < NRBF; ++r) {
        float mu = (2.0f / 15.0f) * (float)r;   // linspace(0, 2, 16)
        float dd = dist - mu;
        o[r] = expf(-dd * dd * 32.0f);          // sigma = 2/16 -> 1/(2s^2)=32
    }
}

// Weight prep: f32 row-major (K x N) -> f16 transposed (N x K), LDS-tiled so
// both global read and write are coalesced.  Per layer this is ~3 M elements;
// it removes all B-side converts/scatter from the hot GEMM loops and halves
// B-side bytes.
__global__ __launch_bounds__(256) void dit_w_t16_kernel(
    const float* __restrict__ W, _Float16* __restrict__ Wt, int K, int N) {
    __shared__ float T[32][33];
    int nt = blockIdx.x * 32, kt = blockIdx.y * 32;
    int c = threadIdx.x & 31, r8 = threadIdx.x >> 5;
    #pragma unroll
    for (int i = 0; i < 4; ++i) {
        int row = r8 + i * 8;
        T[row][c] = W[(size_t)(kt + row) * N + nt + c];
    }
    __syncthreads();
    #pragma unroll
    for (int i = 0; i < 4; ++i) {
        int row = r8 + i * 8;
        Wt[(size_t)(nt + row) * K + kt + c] = (_Float16)T[c][row];
    }
}

// Row LayerNorm over D=512: 128 threads (4 wave32), 4 elems/thread.
__global__ __launch_bounds__(128) void dit_layernorm_kernel(
    const float* __restrict__ X, const float* __restrict__ g,
    const float* __restrict__ b, float* __restrict__ Y) {
    int row = blockIdx.x;
    const float* x = X + (size_t)row * DMODEL;
    int tid = threadIdx.x, lane = tid & 31, wid = tid >> 5;
    float v[4], s = 0.0f, ss = 0.0f;
    #pragma unroll
    for (int i = 0; i < 4; ++i) {
        v[i] = x[tid + i * 128];
        s += v[i];
        ss += v[i] * v[i];
    }
    #pragma unroll
    for (int off = 16; off > 0; off >>= 1) {
        s  += __shfl_xor(s, off, 32);
        ss += __shfl_xor(ss, off, 32);
    }
    __shared__ float sm[4], sm2[4];
    if (lane == 0) { sm[wid] = s; sm2[wid] = ss; }
    __syncthreads();
    s  = sm[0] + sm[1] + sm[2] + sm[3];
    ss = sm2[0] + sm2[1] + sm2[2] + sm2[3];
    float mean = s * (1.0f / DMODEL);
    float var  = ss * (1.0f / DMODEL) - mean * mean;
    float inv  = rsqrtf(var + 1e-5f);
    #pragma unroll
    for (int i = 0; i < 4; ++i) {
        int d = tid + i * 128;
        Y[(size_t)row * DMODEL + d] = (v[i] - mean) * inv * g[d] + b[d];
    }
}

// Row softmax over L=512 (in place).
__global__ __launch_bounds__(128) void dit_softmax_kernel(float* __restrict__ S) {
    int row = blockIdx.x;
    float* x = S + (size_t)row * SEQL;
    int tid = threadIdx.x, lane = tid & 31, wid = tid >> 5;
    float v[4], mx = -3.4e38f;
    #pragma unroll
    for (int i = 0; i < 4; ++i) {
        v[i] = x[tid + i * 128];
        mx = fmaxf(mx, v[i]);
    }
    #pragma unroll
    for (int off = 16; off > 0; off >>= 1) mx = fmaxf(mx, __shfl_xor(mx, off, 32));
    __shared__ float sm[4], sm2[4];
    if (lane == 0) sm[wid] = mx;
    __syncthreads();
    mx = fmaxf(fmaxf(sm[0], sm[1]), fmaxf(sm[2], sm[3]));
    float s = 0.0f;
    #pragma unroll
    for (int i = 0; i < 4; ++i) {
        v[i] = expf(v[i] - mx);
        s += v[i];
    }
    #pragma unroll
    for (int off = 16; off > 0; off >>= 1) s += __shfl_xor(s, off, 32);
    if (lane == 0) sm2[wid] = s;
    __syncthreads();
    s = sm2[0] + sm2[1] + sm2[2] + sm2[3];
    float inv = 1.0f / s;
    #pragma unroll
    for (int i = 0; i < 4; ++i) x[tid + i * 128] = v[i] * inv;
}

// ---------------------------------------------------------------------------
// LDS tiles.  Rows padded to 56 halves (112 B): multiple of 16 B so every
// explicit b64/b128 LDS access is naturally aligned, and 112 B = 28 banks
// gives a conflict-free walk over 16 consecutive rows (28*i mod 64 distinct).
// ---------------------------------------------------------------------------
#define LDSP 56

// Per-lane A fragment per ISA 7.12.2 16-bit A layout (lane=M row; K split
// {0-7,16-23} / {8-15,24-31} between lane halves) -> two b128 LDS reads.
__device__ __forceinline__ v16h dit_load_afrag(const _Float16 (*T)[LDSP],
                                               int row, int lhi) {
    v16h f;
    #pragma unroll
    for (int v = 0; v < 8; ++v) {
        int k = (v < 4) ? ((lhi ? 8 : 0) + 2 * v) : ((lhi ? 24 : 16) + 2 * (v - 4));
        f[2 * v]     = T[row][k];
        f[2 * v + 1] = T[row][k + 1];
    }
    return f;
}

// Per-lane B fragment (lane=N col; K split {0-7,16-23} / {8-15,24-31} between lane halves) -> two b128 reads.
__device__ __forceinline__ v16h dit_load_bfrag(const _Float16 (*T)[LDSP],
                                               int row, int lhi) {
    v16h f;
    #pragma unroll
    for (int v = 0; v < 8; ++v) {
        int k = (v < 4) ? ((lhi ? 8 : 0) + 2 * v) : ((lhi ? 24 : 16) + 2 * (v - 4));
        f[2 * v]     = T[row][k];
        f[2 * v + 1] = T[row][k + 1];
    }
    return f;
}

// ---------------------------------------------------------------------------
// Weight GEMM: A f32 row-major, B f16 pre-transposed [N][K].
// Block = 128 threads (4 wave32) -> 64x64 tile; wave owns 32x32 = 2x2 WMMA
// tiles (4 v_wmma per K-step, 2-way fragment reuse both sides).  LDS is
// double-buffered: next tile's global loads issue before the WMMAs, stash to
// the alternate buffer after -> one barrier per K-step, global latency hidden
// behind matrix ops.  epi: 0 = bias, 1 = bias+gelu, 2 = bias+residual.
// ---------------------------------------------------------------------------
__global__ __launch_bounds__(128) void dit_wmma_gemm_wf16_kernel(
    const float* __restrict__ A, int lda,
    const _Float16* __restrict__ Bt, int ldbT,
    const float* __restrict__ bias,
    const float* Res,
    float* D, int ldd,
    int K, int epi) {
    __shared__ _Float16 As[2][64][LDSP];
    __shared__ _Float16 Bs[2][64][LDSP];
    int m0 = blockIdx.y * 64, n0 = blockIdx.x * 64;
    int tid = threadIdx.x, lane = tid & 31, wid = tid >> 5;
    int wm = wid >> 1, wn = wid & 1;
    int lhi = (lane >= 16) ? 1 : 0;
    int mrow = lane & 15;

    float4 ar[4];
    v8h    br[2];
    auto ldga = [&](int k0) {            // A: 64 rows x 32 k, b128, coalesced on k
        #pragma unroll
        for (int i = 0; i < 4; ++i) {
            int f4 = tid + i * 128;
            int m = f4 >> 3, kq = (f4 & 7) * 4;
            ar[i] = *(const float4*)(A + (size_t)(m0 + m) * lda + k0 + kq);
        }
    };
    auto ldgb = [&](int k0) {            // B: 64 n x 32 k f16, b128, coalesced on k
        #pragma unroll
        for (int i = 0; i < 2; ++i) {
            int p = tid + i * 128;
            int n = p >> 2, kq = (p & 3) * 8;
            br[i] = *(const v8h*)(Bt + (size_t)(n0 + n) * ldbT + k0 + kq);
        }
    };
    auto stash = [&](int buf) {
        #pragma unroll
        for (int i = 0; i < 4; ++i) {
            int f4 = tid + i * 128;
            int m = f4 >> 3, kq = (f4 & 7) * 4;
            v4h hv;
            hv[0] = (_Float16)ar[i].x; hv[1] = (_Float16)ar[i].y;
            hv[2] = (_Float16)ar[i].z; hv[3] = (_Float16)ar[i].w;
            *(v4h*)&As[buf][m][kq] = hv;
        }
        #pragma unroll
        for (int i = 0; i < 2; ++i) {
            int p = tid + i * 128;
            int n = p >> 2, kq = (p & 3) * 8;
            *(v8h*)&Bs[buf][n][kq] = br[i];
        }
    };

    v8f acc00 = {}, acc01 = {}, acc10 = {}, acc11 = {};

    ldga(0); ldgb(0); stash(0);
    __syncthreads();
    int nsteps = K >> 5;
    for (int s = 0; s < nsteps; ++s) {
        int cur = s & 1;
        if (s + 1 < nsteps) { ldga(32 * (s + 1)); ldgb(32 * (s + 1)); }

        v16h a0 = dit_load_afrag(As[cur], wm * 32 + mrow,      lhi);
        v16h a1 = dit_load_afrag(As[cur], wm * 32 + 16 + mrow, lhi);
        v16h b0 = dit_load_bfrag(Bs[cur], wn * 32 + mrow,      lhi);
        v16h b1 = dit_load_bfrag(Bs[cur], wn * 32 + 16 + mrow, lhi);
        acc00 = __builtin_amdgcn_wmma_f32_16x16x32_f16(false, a0, false, b0, (short)0, acc00, false, false);
        acc01 = __builtin_amdgcn_wmma_f32_16x16x32_f16(false, a0, false, b1, (short)0, acc01, false, false);
        acc10 = __builtin_amdgcn_wmma_f32_16x16x32_f16(false, a1, false, b0, (short)0, acc10, false, false);
        acc11 = __builtin_amdgcn_wmma_f32_16x16x32_f16(false, a1, false, b1, (short)0, acc11, false, false);

        if (s + 1 < nsteps) stash(cur ^ 1);
        __syncthreads();
    }

    int gn0 = n0 + wn * 32 + (lane & 15);
    int gn1 = gn0 + 16;
    float bv0 = bias ? bias[gn0] : 0.0f;
    float bv1 = bias ? bias[gn1] : 0.0f;
    const v8f* accs[4] = { &acc00, &acc01, &acc10, &acc11 };
    #pragma unroll
    for (int t = 0; t < 4; ++t) {
        int tm = t >> 1, tn = t & 1;
        int gmb = m0 + wm * 32 + tm * 16 + lhi * 8;   // C/D layout: M = r + 8*lanehi
        int gn  = tn ? gn1 : gn0;
        float bv = tn ? bv1 : bv0;
        #pragma unroll
        for (int r = 0; r < 8; ++r) {
            size_t di = (size_t)(gmb + r) * ldd + gn;
            float v = (*accs[t])[r] + bv;
            if (epi == 1) v = gelu_exact(v);
            else if (epi == 2) v += Res[di];
            D[di] = v;
        }
    }
}

// ---------------------------------------------------------------------------
// Batched f32-B WMMA GEMM (used for P @ V; B is a strided f32 view of the
// packed QKV buffer).  z = zo*zdiv + zi with independent strides.  B-tile
// staging packs k-pairs from two adjacent K rows -> b32 LDS stores.
// ---------------------------------------------------------------------------
__global__ __launch_bounds__(128) void dit_wmma_pv_kernel(
    const float* __restrict__ A, int lda, long long sA1, long long sA0,
    const float* __restrict__ Bm, int ldb, long long sB1, long long sB0,
    float* D, int ldd, long long sD1, long long sD0,
    int K, int zdiv) {
    __shared__ _Float16 As[64][LDSP];
    __shared__ _Float16 Bs[64][LDSP];
    int m0 = blockIdx.y * 64, n0 = blockIdx.x * 64;
    int z = blockIdx.z, zo = z / zdiv, zi = z % zdiv;
    const float* Ap = A + zo * sA1 + zi * sA0;
    const float* Bp = Bm + zo * sB1 + zi * sB0;
    float* Dp = D + zo * sD1 + zi * sD0;

    int tid = threadIdx.x, lane = tid & 31, wid = tid >> 5;
    int wm = wid >> 1, wn = wid & 1;
    int lhi = (lane >= 16) ? 1 : 0;
    int mrow = lane & 15;

    v8f acc00 = {}, acc01 = {}, acc10 = {}, acc11 = {};

    for (int k0 = 0; k0 < K; k0 += 32) {
        #pragma unroll
        for (int i = 0; i < 4; ++i) {          // A: 64 x 32, b128 on k
            int f4 = tid + i * 128;
            int m = f4 >> 3, kq = (f4 & 7) * 4;
            float4 v = *(const float4*)(Ap + (size_t)(m0 + m) * lda + k0 + kq);
            v4h hv;
            hv[0] = (_Float16)v.x; hv[1] = (_Float16)v.y;
            hv[2] = (_Float16)v.z; hv[3] = (_Float16)v.w;
            *(v4h*)&As[m][kq] = hv;
        }
        #pragma unroll
        for (int i = 0; i < 2; ++i) {          // B: 32 k x 64 n, k-pair packed
            int p = tid + i * 128;
            int kk = (p >> 4) * 2, nq = (p & 15) * 4;
            float4 v0 = *(const float4*)(Bp + (size_t)(k0 + kk) * ldb + n0 + nq);
            float4 v1 = *(const float4*)(Bp + (size_t)(k0 + kk + 1) * ldb + n0 + nq);
            v2h p0; p0[0] = (_Float16)v0.x; p0[1] = (_Float16)v1.x; *(v2h*)&Bs[nq + 0][kk] = p0;
            v2h p1; p1[0] = (_Float16)v0.y; p1[1] = (_Float16)v1.y; *(v2h*)&Bs[nq + 1][kk] = p1;
            v2h p2; p2[0] = (_Float16)v0.z; p2[1] = (_Float16)v1.z; *(v2h*)&Bs[nq + 2][kk] = p2;
            v2h p3; p3[0] = (_Float16)v0.w; p3[1] = (_Float16)v1.w; *(v2h*)&Bs[nq + 3][kk] = p3;
        }
        __syncthreads();

        v16h a0 = dit_load_afrag(As, wm * 32 + mrow,      lhi);
        v16h a1 = dit_load_afrag(As, wm * 32 + 16 + mrow, lhi);
        v16h b0 = dit_load_bfrag(Bs, wn * 32 + mrow,      lhi);
        v16h b1 = dit_load_bfrag(Bs, wn * 32 + 16 + mrow, lhi);
        acc00 = __builtin_amdgcn_wmma_f32_16x16x32_f16(false, a0, false, b0, (short)0, acc00, false, false);
        acc01 = __builtin_amdgcn_wmma_f32_16x16x32_f16(false, a0, false, b1, (short)0, acc01, false, false);
        acc10 = __builtin_amdgcn_wmma_f32_16x16x32_f16(false, a1, false, b0, (short)0, acc10, false, false);
        acc11 = __builtin_amdgcn_wmma_f32_16x16x32_f16(false, a1, false, b1, (short)0, acc11, false, false);
        __syncthreads();
    }

    int gn0 = n0 + wn * 32 + (lane & 15);
    int gn1 = gn0 + 16;
    const v8f* accs[4] = { &acc00, &acc01, &acc10, &acc11 };
    #pragma unroll
    for (int t = 0; t < 4; ++t) {
        int tm = t >> 1, tn = t & 1;
        int gmb = m0 + wm * 32 + tm * 16 + lhi * 8;
        int gn  = tn ? gn1 : gn0;
        #pragma unroll
        for (int r = 0; r < 8; ++r)
            Dp[(size_t)(gmb + r) * ldd + gn] = (*accs[t])[r];
    }
}

// ---------------------------------------------------------------------------
// Attention logits: S[z,l,m] = scale*(q.k) + rbf.Wb[:,h] + bb[h], masked.
// 64x64 tiling; K = 64 fully unrolled -> 8 static WMMA; geo bias (float4
// rbf reads) + mask fused in epilogue.
// ---------------------------------------------------------------------------
__global__ __launch_bounds__(128) void dit_attn_logits_kernel(
    const float* __restrict__ qkv, const float* __restrict__ rbf,
    const unsigned char* __restrict__ mask,
    const float* __restrict__ WbL, const float* __restrict__ bbL,
    float* __restrict__ S) {
    __shared__ _Float16 As[64][LDSP];
    __shared__ _Float16 Bs[64][LDSP];
    int l0 = blockIdx.y * 64, m0 = blockIdx.x * 64;
    int z = blockIdx.z, b = z / NHEAD, h = z % NHEAD;
    const int ldq = 3 * DMODEL;
    const float* Q  = qkv + (size_t)b * SEQL * ldq + h * HDIM;            // q slice
    const float* Kp = qkv + (size_t)b * SEQL * ldq + DMODEL + h * HDIM;   // k slice

    int tid = threadIdx.x, lane = tid & 31, wid = tid >> 5;
    int wm = wid >> 1, wn = wid & 1;
    int lhi = (lane >= 16) ? 1 : 0;
    int mrow = lane & 15;

    v8f acc00 = {}, acc01 = {}, acc10 = {}, acc11 = {};

    #pragma unroll
    for (int k0 = 0; k0 < HDIM; k0 += 32) {
        #pragma unroll
        for (int i = 0; i < 4; ++i) {
            int f4 = tid + i * 128;
            int r = f4 >> 3, kq = (f4 & 7) * 4;
            float4 va = *(const float4*)(Q + (size_t)(l0 + r) * ldq + k0 + kq);
            v4h ha;
            ha[0] = (_Float16)va.x; ha[1] = (_Float16)va.y;
            ha[2] = (_Float16)va.z; ha[3] = (_Float16)va.w;
            *(v4h*)&As[r][kq] = ha;
            float4 vb = *(const float4*)(Kp + (size_t)(m0 + r) * ldq + k0 + kq);
            v4h hb;
            hb[0] = (_Float16)vb.x; hb[1] = (_Float16)vb.y;
            hb[2] = (_Float16)vb.z; hb[3] = (_Float16)vb.w;
            *(v4h*)&Bs[r][kq] = hb;     // keys stored n-major, k-contiguous (K^T)
        }
        __syncthreads();

        v16h a0 = dit_load_afrag(As, wm * 32 + mrow,      lhi);
        v16h a1 = dit_load_afrag(As, wm * 32 + 16 + mrow, lhi);
        v16h b0 = dit_load_bfrag(Bs, wn * 32 + mrow,      lhi);
        v16h b1 = dit_load_bfrag(Bs, wn * 32 + 16 + mrow, lhi);
        acc00 = __builtin_amdgcn_wmma_f32_16x16x32_f16(false, a0, false, b0, (short)0, acc00, false, false);
        acc01 = __builtin_amdgcn_wmma_f32_16x16x32_f16(false, a0, false, b1, (short)0, acc01, false, false);
        acc10 = __builtin_amdgcn_wmma_f32_16x16x32_f16(false, a1, false, b0, (short)0, acc10, false, false);
        acc11 = __builtin_amdgcn_wmma_f32_16x16x32_f16(false, a1, false, b1, (short)0, acc11, false, false);
        __syncthreads();
    }

    const float scale = 0.125f;  // 64^-0.5
    float bbh = bbL[h];
    float wcol[NRBF];
    #pragma unroll
    for (int r = 0; r < NRBF; ++r) wcol[r] = WbL[r * NHEAD + h];

    const v8f* accs[4] = { &acc00, &acc01, &acc10, &acc11 };
    #pragma unroll
    for (int t = 0; t < 4; ++t) {
        int tm = t >> 1, tn = t & 1;
        int glb = l0 + wm * 32 + tm * 16 + lhi * 8;
        int gm  = m0 + wn * 32 + tn * 16 + (lane & 15);
        int mok = mask[b * SEQL + gm];
        #pragma unroll
        for (int r = 0; r < 8; ++r) {
            int gl = glb + r;
            float val;
            if (mok) {
                val = (*accs[t])[r] * scale + bbh;
                const float4* rp =
                    (const float4*)(rbf + ((size_t)(b * SEQL + gl) * SEQL + gm) * NRBF);
                #pragma unroll
                for (int q4 = 0; q4 < 4; ++q4) {
                    float4 rv = rp[q4];
                    val = fmaf(rv.x, wcol[4 * q4 + 0], val);
                    val = fmaf(rv.y, wcol[4 * q4 + 1], val);
                    val = fmaf(rv.z, wcol[4 * q4 + 2], val);
                    val = fmaf(rv.w, wcol[4 * q4 + 3], val);
                }
            } else {
                val = -1e30f;
            }
            S[((size_t)z * SEQL + gl) * SEQL + gm] = val;
        }
    }
}

// ---------------------------------------------------------------------------
// Host side
// ---------------------------------------------------------------------------
extern "C" void kernel_launch(void* const* d_in, const int* in_sizes, int n_in,
                              void* d_out, int out_size, void* d_ws, size_t ws_size,
                              hipStream_t stream) {
    (void)in_sizes; (void)n_in; (void)out_size; (void)ws_size;
    const float* x_t  = (const float*)d_in[0];
    const int*   tt   = (const int*)d_in[1];
    const unsigned char* mask = (const unsigned char*)d_in[2];
    const float* xsc  = (const float*)d_in[3];
    const float* Wi   = (const float*)d_in[4];
    const float* bi   = (const float*)d_in[5];
    const float* Wsc  = (const float*)d_in[6];
    const float* bsc  = (const float*)d_in[7];
    const float* Wt1  = (const float*)d_in[8];
    const float* bt1  = (const float*)d_in[9];
    const float* Wt2  = (const float*)d_in[10];
    const float* bt2  = (const float*)d_in[11];
    const float* g1   = (const float*)d_in[12];
    const float* b1   = (const float*)d_in[13];
    const float* Wqkv = (const float*)d_in[14];
    const float* bqkv = (const float*)d_in[15];
    const float* Wo   = (const float*)d_in[16];
    const float* bo   = (const float*)d_in[17];
    const float* Wb   = (const float*)d_in[18];
    const float* bb   = (const float*)d_in[19];
    const float* g2   = (const float*)d_in[20];
    const float* b2   = (const float*)d_in[21];
    const float* Wf1  = (const float*)d_in[22];
    const float* bf1  = (const float*)d_in[23];
    const float* Wf2  = (const float*)d_in[24];
    const float* bf2  = (const float*)d_in[25];
    const float* gfin = (const float*)d_in[26];
    const float* bfin = (const float*)d_in[27];
    const float* Wout = (const float*)d_in[28];
    const float* bout = (const float*)d_in[29];

    char* wsp = (char*)d_ws;
    size_t off = 0;
    auto alloc = [&](size_t bytes) -> void* {
        void* p = (void*)(wsp + off);
        off += (bytes + 255) & ~(size_t)255;
        return p;
    };
    float* h    = (float*)alloc((size_t)BATCH * SEQL * DMODEL * 4);
    float* xln  = (float*)alloc((size_t)BATCH * SEQL * DMODEL * 4);
    float* qkv  = (float*)alloc((size_t)BATCH * SEQL * 3 * DMODEL * 4);
    float* obuf = (float*)alloc((size_t)BATCH * SEQL * DMODEL * 4);
    float* f1   = (float*)alloc((size_t)BATCH * SEQL * DFFN * 4);
    float* S    = (float*)alloc((size_t)BATCH * NHEAD * SEQL * SEQL * 4);
    float* rbf  = (float*)alloc((size_t)BATCH * SEQL * SEQL * NRBF * 4);
    float* t0   = (float*)alloc((size_t)BATCH * DMODEL * 4);
    float* t1   = (float*)alloc((size_t)BATCH * DMODEL * 4);
    float* t2   = (float*)alloc((size_t)BATCH * DMODEL * 4);
    // per-layer transposed f16 weights (regenerated each layer iteration)
    _Float16* qkvT = (_Float16*)alloc((size_t)3 * DMODEL * DMODEL * 2);
    _Float16* oT   = (_Float16*)alloc((size_t)DMODEL * DMODEL * 2);
    _Float16* f1T  = (_Float16*)alloc((size_t)DFFN * DMODEL * 2);
    _Float16* f2T  = (_Float16*)alloc((size_t)DMODEL * DFFN * 2);

    // --- preamble ---
    dit_sinus_t_kernel<<<(BATCH * DMODEL + 255) / 256, 256, 0, stream>>>(tt, t0);
    dit_plain_gemm_kernel<<<(BATCH * DMODEL + 255) / 256, 256, 0, stream>>>(
        t0, Wt1, bt1, t1, BATCH, DMODEL, DMODEL, 1);
    dit_plain_gemm_kernel<<<(BATCH * DMODEL + 255) / 256, 256, 0, stream>>>(
        t1, Wt2, bt2, t2, BATCH, DMODEL, DMODEL, 0);
    dit_embed_kernel<<<(BATCH * SEQL * DMODEL + 255) / 256, 256, 0, stream>>>(
        x_t, xsc, Wi, bi, Wsc, bsc, t2, h);
    dit_rbf_kernel<<<(BATCH * SEQL * SEQL + 255) / 256, 256, 0, stream>>>(x_t, rbf);

    const int M = BATCH * SEQL;  // 2048 token rows

    for (int l = 0; l < NLAYER; ++l) {
        const float* g1l = g1 + l * DMODEL;
        const float* b1l = b1 + l * DMODEL;
        const float* Wqkvl = Wqkv + (size_t)l * DMODEL * 3 * DMODEL;
        const float* bqkvl = bqkv + (size_t)l * 3 * DMODEL;
        const float* Wol = Wo + (size_t)l * DMODEL * DMODEL;
        const float* bol = bo + (size_t)l * DMODEL;
        const float* Wbl = Wb + (size_t)l * NRBF * NHEAD;
        const float* bbl = bb + (size_t)l * NHEAD;
        const float* g2l = g2 + l * DMODEL;
        const float* b2l = b2 + l * DMODEL;
        const float* Wf1l = Wf1 + (size_t)l * DMODEL * DFFN;
        const float* bf1l = bf1 + (size_t)l * DFFN;
        const float* Wf2l = Wf2 + (size_t)l * DFFN * DMODEL;
        const float* bf2l = bf2 + (size_t)l * DMODEL;

        // weight prep: f32 (K x N) -> f16 (N x K)
        { dim3 g(3 * DMODEL / 32, DMODEL / 32); dit_w_t16_kernel<<<g, 256, 0, stream>>>(Wqkvl, qkvT, DMODEL, 3 * DMODEL); }
        { dim3 g(DMODEL / 32, DMODEL / 32);     dit_w_t16_kernel<<<g, 256, 0, stream>>>(Wol,   oT,   DMODEL, DMODEL); }
        { dim3 g(DFFN / 32, DMODEL / 32);       dit_w_t16_kernel<<<g, 256, 0, stream>>>(Wf1l,  f1T,  DMODEL, DFFN); }
        { dim3 g(DMODEL / 32, DFFN / 32);       dit_w_t16_kernel<<<g, 256, 0, stream>>>(Wf2l,  f2T,  DFFN, DMODEL); }

        // x = LN(h)
        dit_layernorm_kernel<<<M, 128, 0, stream>>>(h, g1l, b1l, xln);

        // qkv = x @ Wqkv + bqkv   (2048 x 1536 x 512)
        {
            dim3 g(3 * DMODEL / 64, M / 64);
            dit_wmma_gemm_wf16_kernel<<<g, 128, 0, stream>>>(
                xln, DMODEL, qkvT, DMODEL, bqkvl, nullptr, qkv, 3 * DMODEL,
                DMODEL, 0);
        }

        // logits + geo bias + mask
        {
            dim3 g(SEQL / 64, SEQL / 64, BATCH * NHEAD);
            dit_attn_logits_kernel<<<g, 128, 0, stream>>>(qkv, rbf, mask, Wbl, bbl, S);
        }

        dit_softmax_kernel<<<BATCH * NHEAD * SEQL, 128, 0, stream>>>(S);

        // o = P @ V : batched over (b,h); V is a strided view of qkv
        {
            dim3 g(HDIM / 64, SEQL / 64, BATCH * NHEAD);
            dit_wmma_pv_kernel<<<g, 128, 0, stream>>>(
                S, SEQL, (long long)NHEAD * SEQL * SEQL, (long long)SEQL * SEQL,
                qkv + 2 * DMODEL, 3 * DMODEL, (long long)SEQL * 3 * DMODEL, HDIM,
                obuf, DMODEL, (long long)SEQL * DMODEL, HDIM,
                SEQL, NHEAD);
        }

        // h = h + o @ Wo + bo
        {
            dim3 g(DMODEL / 64, M / 64);
            dit_wmma_gemm_wf16_kernel<<<g, 128, 0, stream>>>(
                obuf, DMODEL, oT, DMODEL, bol, h, h, DMODEL, DMODEL, 2);
        }

        // x2 = LN(h); f1 = gelu(x2 @ Wf1 + bf1); h = h + f1 @ Wf2 + bf2
        dit_layernorm_kernel<<<M, 128, 0, stream>>>(h, g2l, b2l, xln);
        {
            dim3 g(DFFN / 64, M / 64);
            dit_wmma_gemm_wf16_kernel<<<g, 128, 0, stream>>>(
                xln, DMODEL, f1T, DMODEL, bf1l, nullptr, f1, DFFN, DMODEL, 1);
        }
        {
            dim3 g(DMODEL / 64, M / 64);
            dit_wmma_gemm_wf16_kernel<<<g, 128, 0, stream>>>(
                f1, DFFN, f2T, DFFN, bf2l, h, h, DMODEL, DFFN, 2);
        }
    }

    // out = LN(h) @ Wout + bout  (N=3 -> plain kernel)
    dit_layernorm_kernel<<<M, 128, 0, stream>>>(h, gfin, bfin, xln);
    dit_plain_gemm_kernel<<<(M * 3 + 255) / 256, 256, 0, stream>>>(
        xln, Wout, bout, (float*)d_out, M, 3, DMODEL, 0);
}
